// FFTBlock_89421219103445
// MI455X (gfx1250) — matrix operand order repaired and checked by reading, not compiled, not run on hardware
//
#include <hip/hip_runtime.h>
#include <hip/hip_bf16.h>
#include <cstdint>

#define BB  16
#define SS  1024
#define DD  512
#define HH  8
#define DKK 64
#define CDD 2048
#define KSS 9

typedef __attribute__((ext_vector_type(16))) __bf16 v16bf;
typedef __attribute__((ext_vector_type(8)))  __bf16 v8bf;
typedef __attribute__((ext_vector_type(8)))  float  v8f;

static __device__ __forceinline__ v8f wmma_bf16(v16bf a, v16bf b, v8f c) {
  // (neg_a, A, neg_b, B, c_mod, C, reuse_a, reuse_b)
  return __builtin_amdgcn_wmma_f32_16x16x32_bf16(false, a, false, b, (short)0, c, false, false);
}

// A-operand (16x32, M x K), row-major source with leading dim ld (elements).
// Lane: row = caller-provided absolute row; element j -> K = kb+16*half+j (32B contiguous).
static __device__ __forceinline__ v16bf load_a(const __bf16* p, int ld, int row, int kb, int hf) {
  return *(const v16bf*)(p + (size_t)row * ld + kb + 16 * hf);
}

// B-operand (32x16, K x N) read from a [N][K] (pre-transposed) matrix:
// lane col = caller-provided; element j -> K = kb + 16*half + j (32B contiguous).
static __device__ __forceinline__ v16bf load_b(const __bf16* p, int ld, int col, int kb, int hf) {
  return *(const v16bf*)(p + (size_t)col * ld + kb + 16 * hf);
}

static __device__ __forceinline__ v16bf zero16() {
  v16bf z;
#pragma unroll
  for (int j = 0; j < 16; ++j) z[j] = (__bf16)0.0f;
  return z;
}

// ---------------- elementwise convert / repack ----------------
__global__ void k_cvt_bf16(const float* __restrict__ src, __bf16* __restrict__ dst, int n) {
  int i = blockIdx.x * blockDim.x + threadIdx.x;
  if (i < n) dst[i] = (__bf16)src[i];
}

// Wq [H, D, DK] (f32) -> WT [H, DK, D] (bf16): contiguous in D for B-operand loads.
__global__ void k_repack_qkv(const float* __restrict__ W, __bf16* __restrict__ WT) {
  int idx = blockIdx.x * blockDim.x + threadIdx.x;
  if (idx >= HH * DD * DKK) return;
  int dk = idx % DKK;
  int d  = (idx / DKK) % DD;
  int h  = idx / (DKK * DD);
  WT[((size_t)h * DKK + dk) * DD + d] = (__bf16)W[idx];
}

// Wo [K=512, N=512] -> WoT [N][K]
__global__ void k_repack_wo(const float* __restrict__ W, __bf16* __restrict__ WT) {
  int idx = blockIdx.x * blockDim.x + threadIdx.x;
  if (idx >= DD * DD) return;
  int n = idx % DD;
  int k = idx / DD;
  WT[(size_t)n * DD + k] = (__bf16)W[idx];
}

// Wc [O, I, 9] (OIH) -> Wp [9][O][I]: per-tap weight matrix, contiguous in I.
__global__ void k_repack_conv(const float* __restrict__ W, __bf16* __restrict__ Wp, int O, int I) {
  int idx = blockIdx.x * blockDim.x + threadIdx.x;
  if (idx >= O * I * KSS) return;
  int kk = idx % KSS;
  int i  = (idx / KSS) % I;
  int o  = idx / (KSS * I);
  Wp[((size_t)kk * O + o) * I + i] = (__bf16)W[idx];
}

// ---------------- QKV projection (one wave -> 16 s x 16 dk tile) ----------------
__global__ void __launch_bounds__(32) k_qkv(
    const __bf16* __restrict__ xbf,
    const __bf16* __restrict__ wqT, const __bf16* __restrict__ wkT, const __bf16* __restrict__ wvT,
    const float* __restrict__ bq, const float* __restrict__ bk, const float* __restrict__ bv,
    __bf16* __restrict__ qout, __bf16* __restrict__ kout, __bf16* __restrict__ vtout) {
  int lane = threadIdx.x, l16 = lane & 15, hf = lane >> 4;
  int b = blockIdx.z / HH, h = blockIdx.z % HH;
  int sT = blockIdx.x * 16;
  int which = blockIdx.y >> 2;   // 0=Q 1=K 2=V
  int nb    = blockIdx.y & 3;    // dk block
  const __bf16* WT   = (which == 0 ? wqT : (which == 1 ? wkT : wvT)) + (size_t)h * DKK * DD;
  const float*  bias = (which == 0 ? bq  : (which == 1 ? bk  : bv )) + h * DKK;
  const __bf16* Ap = xbf + ((size_t)b * SS + sT) * DD;

  v8f acc = {};
  for (int kb = 0; kb < DD; kb += 32) {
    v16bf a  = load_a(Ap, DD, l16, kb, hf);
    v16bf bm = load_b(WT, DD, nb * 16 + l16, kb, hf);
    acc = wmma_bf16(a, bm, acc);
  }
  int   dk    = nb * 16 + l16;
  float bvv   = bias[dk];
  float scale = (which == 0) ? 0.125f : 1.0f;  // fold 1/sqrt(64) into Q
  if (which < 2) {
    __bf16* dst = (which == 0) ? qout : kout;  // [B,H,S,DK]
#pragma unroll
    for (int r = 0; r < 8; ++r) {
      int s = sT + 8 * hf + r;
      dst[(((size_t)b * HH + h) * SS + s) * DKK + dk] = (__bf16)((acc[r] + bvv) * scale);
    }
  } else {                                      // V transposed: [B,H,DK,S]
#pragma unroll
    for (int r = 0; r < 8; ++r) {
      int s = sT + 8 * hf + r;
      vtout[(((size_t)b * HH + h) * DKK + dk) * SS + s] = (__bf16)(acc[r] + bvv);
    }
  }
}

// ---------------- attention: one wave per (b, h, 16-row s tile) ----------------
__global__ void __launch_bounds__(32) k_attn(
    const __bf16* __restrict__ qbuf, const __bf16* __restrict__ kbuf,
    const __bf16* __restrict__ vtbuf, __bf16* __restrict__ ctxb) {
  __shared__ float sc[16 * SS];  // 64KB score strip
  int lane = threadIdx.x, l16 = lane & 15, hf = lane >> 4;
  int b = blockIdx.z, h = blockIdx.y, sT = blockIdx.x * 16;
  const __bf16* Qp = qbuf  + (((size_t)b * HH + h) * SS + sT) * DKK;
  const __bf16* Kp = kbuf  + ((size_t)b * HH + h) * SS * DKK;
  const __bf16* Vp = vtbuf + ((size_t)b * HH + h) * DKK * SS;

  v16bf qa0 = load_a(Qp, DKK, l16, 0, hf);
  v16bf qa1 = load_a(Qp, DKK, l16, 32, hf);

  // pass 1: scores -> LDS
  for (int tb = 0; tb < SS; tb += 16) {
    v16bf kb0 = load_b(Kp, DKK, tb + l16, 0, hf);
    v16bf kb1 = load_b(Kp, DKK, tb + l16, 32, hf);
    v8f s = {};
    s = wmma_bf16(qa0, kb0, s);
    s = wmma_bf16(qa1, kb1, s);
#pragma unroll
    for (int r = 0; r < 8; ++r) sc[(8 * hf + r) * SS + tb + l16] = s[r];
  }
  __syncthreads();

  // row stats: lanes L and L+16 split row l16
  int   row = l16;
  float mx  = -3.0e38f;
  for (int t = hf * 512; t < hf * 512 + 512; ++t) mx = fmaxf(mx, sc[row * SS + t]);
  mx = fmaxf(mx, __shfl_xor(mx, 16, 32));
  float sum = 0.f;
  for (int t = hf * 512; t < hf * 512 + 512; ++t) sum += __expf(sc[row * SS + t] - mx);
  sum += __shfl_xor(sum, 16, 32);

  // pass 2: ctx = exp(S - m) @ V ; probs built directly in A-operand layout
  v8f acc0 = {}, acc1 = {}, acc2 = {}, acc3 = {};
  for (int tb = 0; tb < SS; tb += 32) {
    v16bf pa;
#pragma unroll
    for (int j = 0; j < 16; ++j) {
      int t = tb + 16 * hf + j;
      pa[j] = (__bf16)__expf(sc[row * SS + t] - mx);
    }
    acc0 = wmma_bf16(pa, load_b(Vp, SS,  0 + l16, tb, hf), acc0);
    acc1 = wmma_bf16(pa, load_b(Vp, SS, 16 + l16, tb, hf), acc1);
    acc2 = wmma_bf16(pa, load_b(Vp, SS, 32 + l16, tb, hf), acc2);
    acc3 = wmma_bf16(pa, load_b(Vp, SS, 48 + l16, tb, hf), acc3);
  }
  float inv[8];
#pragma unroll
  for (int r = 0; r < 8; ++r) inv[r] = 1.0f / __shfl(sum, 8 * hf + r, 32);
#pragma unroll
  for (int r = 0; r < 8; ++r) {
    int    s    = sT + 8 * hf + r;
    size_t base = ((size_t)b * SS + s) * (HH * DKK) + (size_t)h * DKK;
    ctxb[base +  0 + l16] = (__bf16)(acc0[r] * inv[r]);
    ctxb[base + 16 + l16] = (__bf16)(acc1[r] * inv[r]);
    ctxb[base + 32 + l16] = (__bf16)(acc2[r] * inv[r]);
    ctxb[base + 48 + l16] = (__bf16)(acc3[r] * inv[r]);
  }
}

// ---------------- output projection: ctx[BS,512] @ WoT -> attn f32 ----------------
// 4x4 register-blocked: one wave computes a 64(row) x 64(col) macro-tile.
__global__ void __launch_bounds__(32) k_gemm_wo(
    const __bf16* __restrict__ ctxb, const __bf16* __restrict__ woT, float* __restrict__ attnb) {
  int lane = threadIdx.x, l16 = lane & 15, hf = lane >> 4;
  int rT = blockIdx.x * 64, nT = blockIdx.y * 64;
  v8f acc[4][4] = {};
  for (int kb = 0; kb < DD; kb += 32) {
    v16bf a[4];
#pragma unroll
    for (int rb = 0; rb < 4; ++rb)
      a[rb] = load_a(ctxb, DD, rT + rb * 16 + l16, kb, hf);
#pragma unroll
    for (int nb = 0; nb < 4; ++nb) {
      v16bf bm = load_b(woT, DD, nT + nb * 16 + l16, kb, hf);
#pragma unroll
      for (int rb = 0; rb < 4; ++rb)
        acc[rb][nb] = wmma_bf16(a[rb], bm, acc[rb][nb]);
    }
  }
#pragma unroll
  for (int rb = 0; rb < 4; ++rb)
#pragma unroll
    for (int nb = 0; nb < 4; ++nb)
#pragma unroll
      for (int r = 0; r < 8; ++r)
        attnb[(size_t)(rT + rb * 16 + 8 * hf + r) * DD + nT + nb * 16 + l16] = acc[rb][nb][r];
}

// ---------------- LayerNorm 1: h = LN(x + attn + bo) ----------------
__global__ void __launch_bounds__(32) k_ln1(
    const float* __restrict__ x, const float* __restrict__ attnb, const float* __restrict__ bo,
    const float* __restrict__ g, const float* __restrict__ bb,
    float* __restrict__ hbuf, __bf16* __restrict__ hbf) {
  int row = blockIdx.x, lane = threadIdx.x;
  const float* xr = x + (size_t)row * DD;
  const float* ar = attnb + (size_t)row * DD;
  float v[16], s = 0.f, s2 = 0.f;
#pragma unroll
  for (int i = 0; i < 16; ++i) {
    int   d = lane + i * 32;
    float t = xr[d] + ar[d] + bo[d];
    v[i] = t; s += t; s2 += t * t;
  }
#pragma unroll
  for (int o = 16; o > 0; o >>= 1) { s += __shfl_xor(s, o, 32); s2 += __shfl_xor(s2, o, 32); }
  float mean = s * (1.0f / DD);
  float var  = s2 * (1.0f / DD) - mean * mean;
  float rstd = rsqrtf(var + 1e-5f);
#pragma unroll
  for (int i = 0; i < 16; ++i) {
    int   d = lane + i * 32;
    float o = (v[i] - mean) * rstd * g[d] + bb[d];
    hbuf[(size_t)row * DD + d] = o;
    hbf[(size_t)row * DD + d]  = (__bf16)o;
  }
}

// ---------------- conv1d ('same', k=9) as implicit GEMM ----------------
// 4x4 register-blocked: one wave computes a 64(ch) x 64(s) macro-tile.
// in: [B, S, CI] bf16 (channel-last). Wp: [9][CO][CI] bf16. out: [B, S, CO].
template <int CI, int CO, bool OUTBF>
__global__ void __launch_bounds__(32) k_conv(
    const __bf16* __restrict__ in, const __bf16* __restrict__ Wp,
    const float* __restrict__ bias, void* __restrict__ outv) {
  int lane = threadIdx.x, l16 = lane & 15, hf = lane >> 4;
  int b = blockIdx.z, oT = blockIdx.y * 64, sT = blockIdx.x * 64;
  v8f acc[4][4] = {};  // [ob][sb]
  for (int kk = 0; kk < KSS; ++kk) {
    const __bf16* Wk = Wp + (size_t)kk * CO * CI;
    // warm L2 for this tap's weight panel (global_prefetch_b8)
    __builtin_prefetch(Wk + (size_t)(oT + l16) * CI, 0, 1);
    bool ok[4];
    const __bf16* inrow[4];
#pragma unroll
    for (int sb = 0; sb < 4; ++sb) {
      int sp = sT + sb * 16 + l16 + kk - 4;  // this lane's input s position
      ok[sb] = (sp >= 0) && (sp < SS);
      inrow[sb] = in + ((size_t)b * SS + (ok[sb] ? sp : 0)) * CI;
    }
    for (int ib = 0; ib < CI; ib += 32) {
      v16bf a[4];
#pragma unroll
      for (int ob = 0; ob < 4; ++ob)
        a[ob] = load_a(Wk, CI, oT + ob * 16 + l16, ib, hf);
#pragma unroll
      for (int sb = 0; sb < 4; ++sb) {
        v16bf bm = *(const v16bf*)(inrow[sb] + ib + 16 * hf);
        if (!ok[sb]) bm = zero16();          // zero-pad boundary columns
#pragma unroll
        for (int ob = 0; ob < 4; ++ob)
          acc[ob][sb] = wmma_bf16(a[ob], bm, acc[ob][sb]);
      }
    }
  }
#pragma unroll
  for (int ob = 0; ob < 4; ++ob) {
#pragma unroll
    for (int r = 0; r < 8; ++r) {
      int   ch = oT + ob * 16 + 8 * hf + r;
      float bv = bias[ch];
#pragma unroll
      for (int sb = 0; sb < 4; ++sb) {
        float  val = fmaxf(acc[ob][sb][r] + bv, 0.0f);
        size_t idx = ((size_t)b * SS + sT + sb * 16 + l16) * CO + ch;
        if constexpr (OUTBF) ((__bf16*)outv)[idx] = (__bf16)val;
        else                 ((float*)outv)[idx]  = val;
      }
    }
  }
}

// ---------------- LayerNorm 2: out = LN(h + conv_out) ----------------
__global__ void __launch_bounds__(32) k_ln2(
    const float* __restrict__ hbuf, const float* __restrict__ c2,
    const float* __restrict__ g, const float* __restrict__ bb, float* __restrict__ out) {
  int row = blockIdx.x, lane = threadIdx.x;
  const float* hr = hbuf + (size_t)row * DD;
  const float* cr = c2 + (size_t)row * DD;
  float v[16], s = 0.f, s2 = 0.f;
#pragma unroll
  for (int i = 0; i < 16; ++i) {
    int   d = lane + i * 32;
    float t = hr[d] + cr[d];
    v[i] = t; s += t; s2 += t * t;
  }
#pragma unroll
  for (int o = 16; o > 0; o >>= 1) { s += __shfl_xor(s, o, 32); s2 += __shfl_xor(s2, o, 32); }
  float mean = s * (1.0f / DD);
  float var  = s2 * (1.0f / DD) - mean * mean;
  float rstd = rsqrtf(var + 1e-5f);
#pragma unroll
  for (int i = 0; i < 16; ++i) {
    int d = lane + i * 32;
    out[(size_t)row * DD + d] = (v[i] - mean) * rstd * g[d] + bb[d];
  }
}

extern "C" void kernel_launch(void* const* d_in, const int* in_sizes, int n_in,
                              void* d_out, int out_size, void* d_ws, size_t ws_size,
                              hipStream_t stream) {
  (void)in_sizes; (void)n_in; (void)out_size; (void)ws_size;
  const float* x   = (const float*)d_in[0];
  const float* Wq  = (const float*)d_in[1];
  const float* bq  = (const float*)d_in[2];
  const float* Wk  = (const float*)d_in[3];
  const float* bk  = (const float*)d_in[4];
  const float* Wv  = (const float*)d_in[5];
  const float* bv  = (const float*)d_in[6];
  const float* Wo  = (const float*)d_in[7];
  const float* bo  = (const float*)d_in[8];
  const float* g1  = (const float*)d_in[9];
  const float* b1  = (const float*)d_in[10];
  const float* g2  = (const float*)d_in[11];
  const float* b2  = (const float*)d_in[12];
  const float* Wc1 = (const float*)d_in[13];
  const float* bc1 = (const float*)d_in[14];
  const float* Wc2 = (const float*)d_in[15];
  const float* bc2 = (const float*)d_in[16];
  float* out = (float*)d_out;

  char*  ws  = (char*)d_ws;
  size_t off = 0;
  auto take = [&](size_t bytes) -> char* {
    char* p = ws + off;
    off += (bytes + 255) & ~(size_t)255;
    return p;
  };
  __bf16* xbf   = (__bf16*)take((size_t)BB * SS * DD * 2);
  __bf16* wqT   = (__bf16*)take((size_t)HH * DKK * DD * 2);
  __bf16* wkT   = (__bf16*)take((size_t)HH * DKK * DD * 2);
  __bf16* wvT   = (__bf16*)take((size_t)HH * DKK * DD * 2);
  __bf16* woT   = (__bf16*)take((size_t)DD * DD * 2);
  __bf16* wc1p  = (__bf16*)take((size_t)KSS * CDD * DD * 2);
  __bf16* wc2p  = (__bf16*)take((size_t)KSS * DD * CDD * 2);
  __bf16* qbuf  = (__bf16*)take((size_t)BB * HH * SS * DKK * 2);
  __bf16* kbuf  = (__bf16*)take((size_t)BB * HH * SS * DKK * 2);
  __bf16* vtbuf = (__bf16*)take((size_t)BB * HH * DKK * SS * 2);
  __bf16* ctxb  = (__bf16*)take((size_t)BB * SS * DD * 2);
  float*  attnb = (float*)take((size_t)BB * SS * DD * 4);
  float*  hbuf  = (float*)take((size_t)BB * SS * DD * 4);
  __bf16* hbf   = (__bf16*)take((size_t)BB * SS * DD * 2);
  // reuse: Q/K/Vt/ctx (64MB) consumed before conv1 writes; attn consumed before conv2 writes
  __bf16* c1buf = qbuf;
  float*  c2buf = attnb;

  // phase 0: convert / repack
  {
    int n = BB * SS * DD;
    k_cvt_bf16<<<(n + 255) / 256, 256, 0, stream>>>(x, xbf, n);
  }
  {
    int n = HH * DD * DKK;
    k_repack_qkv<<<(n + 255) / 256, 256, 0, stream>>>(Wq, wqT);
    k_repack_qkv<<<(n + 255) / 256, 256, 0, stream>>>(Wk, wkT);
    k_repack_qkv<<<(n + 255) / 256, 256, 0, stream>>>(Wv, wvT);
  }
  {
    int n = DD * DD;
    k_repack_wo<<<(n + 255) / 256, 256, 0, stream>>>(Wo, woT);
  }
  {
    int n = CDD * DD * KSS;
    k_repack_conv<<<(n + 255) / 256, 256, 0, stream>>>(Wc1, wc1p, CDD, DD);
    k_repack_conv<<<(n + 255) / 256, 256, 0, stream>>>(Wc2, wc2p, DD, CDD);
  }

  // phase 1: QKV projections (Q pre-scaled by 1/8)
  k_qkv<<<dim3(SS / 16, 12, BB * HH), 32, 0, stream>>>(xbf, wqT, wkT, wvT, bq, bk, bv,
                                                       qbuf, kbuf, vtbuf);
  // phase 2: attention
  k_attn<<<dim3(SS / 16, HH, BB), 32, 0, stream>>>(qbuf, kbuf, vtbuf, ctxb);
  // phase 3: output projection (64x64 macro-tiles)
  k_gemm_wo<<<dim3(BB * SS / 64, DD / 64), 32, 0, stream>>>(ctxb, woT, attnb);
  // phase 4: LN1 -> h (f32 + bf16)
  k_ln1<<<BB * SS, 32, 0, stream>>>(x, attnb, bo, g1, b1, hbuf, hbf);
  // phase 5: conv FFN (64x64 macro-tiles)
  k_conv<DD, CDD, true><<<dim3(SS / 64, CDD / 64, BB), 32, 0, stream>>>(hbf, wc1p, bc1, (void*)c1buf);
  k_conv<CDD, DD, false><<<dim3(SS / 64, DD / 64, BB), 32, 0, stream>>>(c1buf, wc2p, bc2, (void*)c2buf);
  // phase 6: LN2 -> out
  k_ln2<<<BB * SS, 32, 0, stream>>>(hbuf, c2buf, g2, b2, out);
}
